// DeformableSkipBlock_57466662420730
// MI455X (gfx1250) — matrix operand order repaired and checked by reading, not compiled, not run on hardware
//
#include <hip/hip_runtime.h>
#include <hip/hip_bf16.h>

typedef __attribute__((ext_vector_type(16))) __bf16 v16bf;
typedef __attribute__((ext_vector_type(8)))  float  v8f;
typedef unsigned short ushort_t;

#define CIN  256
#define CH   256
#define HW   4096
#define WID  64

// round-to-nearest-even float -> bf16 (bit-level)
static __device__ __forceinline__ ushort_t f2bf(float f) {
    unsigned u = __builtin_bit_cast(unsigned, f);
    u = u + 0x7FFFu + ((u >> 16) & 1u);
    return (ushort_t)(u >> 16);
}

// ---------------------------------------------------------------------------
// Repack w_def (Cout, Cin, 3, 3) fp32 into the per-lane WMMA A-fragment bf16
// layout: flat[k][chunk][tile][lane][16 halves], one contiguous v16bf per
// (tile,lane).  16-bit A 16x32: lane L -> M = L%16,
// Khalf = (L/16)*16; element s: K = Khalf + s (contiguous, matches B).
// ---------------------------------------------------------------------------
__global__ __launch_bounds__(256) void prep_weight_kernel(
    const float* __restrict__ w, ushort_t* __restrict__ wbf)
{
    int f     = blockIdx.x * 256 + threadIdx.x;     // grid = 2304 blocks
    int s     = f & 15;
    int lane  = (f >> 4) & 31;
    int tile  = (f >> 9) & 15;
    int chunk = (f >> 13) & 7;
    int k     = f >> 16;                            // 0..8
    int v     = s >> 1, h = s & 1;
    int khalf = (lane >> 4) << 4;
    int kloc  = khalf + s;
    int cin   = chunk * 32 + kloc;
    int cout  = tile * 16 + (lane & 15);
    wbf[f] = f2bf(w[((size_t)cout * CIN + cin) * 9 + k]);
}

// ---------------------------------------------------------------------------
// offset = conv3x3(x, w_off) + b_off.  One thread per pixel, all 18 output
// channels in registers (x patch loaded once, not 18x).  Weights staged
// through LDS in 32-cin chunks (18*32*9*4B = 20.7 KB).
// grid = B*H*W/256 = 64 blocks.
// ---------------------------------------------------------------------------
__global__ __launch_bounds__(256) void offset_conv_kernel(
    const float* __restrict__ x, const float* __restrict__ w,
    const float* __restrict__ bias, float* __restrict__ out)
{
    __shared__ float sw[18 * 32 * 9];               // [oc][cl][k]

    int p   = blockIdx.x * 256 + threadIdx.x;
    int b   = p >> 12;
    int pix = p & (HW - 1);
    int yy  = pix >> 6, xx = pix & 63;

    float acc[18];
#pragma unroll
    for (int oc = 0; oc < 18; ++oc) acc[oc] = bias[oc];

    const float* xb = x + (size_t)b * CIN * HW;

    for (int c0 = 0; c0 < CIN; c0 += 32) {
        __syncthreads();
        for (int i = threadIdx.x; i < 18 * 32 * 9; i += 256) {
            int oc = i / 288, rem = i % 288;        // rem = cl*9 + k
            sw[i] = w[(size_t)oc * (CIN * 9) + c0 * 9 + rem];
        }
        __syncthreads();

        for (int cl = 0; cl < 32; ++cl) {
            const float* xp = xb + (size_t)(c0 + cl) * HW;
            float xv[9];
#pragma unroll
            for (int ky = 0; ky < 3; ++ky) {
#pragma unroll
                for (int kx = 0; kx < 3; ++kx) {
                    int sy = yy + ky - 1, sx = xx + kx - 1;
                    bool ok = (sy >= 0) & (sy < 64) & (sx >= 0) & (sx < 64);
                    xv[ky * 3 + kx] = ok ? xp[sy * WID + sx] : 0.0f;
                }
            }
#pragma unroll
            for (int oc = 0; oc < 18; ++oc) {
                const float* wr = &sw[oc * 288 + cl * 9];
#pragma unroll
                for (int t = 0; t < 9; ++t)
                    acc[oc] = fmaf(wr[t], xv[t], acc[oc]);
            }
        }
    }
#pragma unroll
    for (int oc = 0; oc < 18; ++oc)
        out[(((size_t)b * 18 + oc) * HW) + pix] = acc[oc];
}

// ---------------------------------------------------------------------------
// Deformable conv core on v_wmma_f32_16x16x32_bf16.
// Workgroup = 8 wave32, handles one (b, 32-pixel tile) x all 256 couts.
// Each wave: 2 M-tiles x 2 N-tiles -> 4 accumulators, 4 WMMAs per K-chunk.
// Gather lanes sweep adjacent PIXELS (coalesced: neighbors sample adjacent
// addresses) instead of cin planes.  Ping-pong B tile: 1 barrier per chunk.
// ---------------------------------------------------------------------------
__global__ __launch_bounds__(256) void deform_wmma_kernel(
    const float* __restrict__ x, const float* __restrict__ offs,
    const ushort_t* __restrict__ wbf, float* __restrict__ out)
{
    // row stride 48 ushorts = 96B: keeps 32B alignment for v16bf reads,
    // staggers banks for the stride-row ds_load_b128 pattern.
    __shared__ __align__(64) ushort_t valB[2][32][48];
    __shared__ int   sIdx[32][4];
    __shared__ float sWgt[32][4];

    const int tid  = threadIdx.x;
    const int lane = tid & 31;
    const int wave = tid >> 5;
    const int b    = blockIdx.x >> 7;           // grid = 512
    const int pix0 = (blockIdx.x & 127) << 5;   // 32 consecutive pixels (half row)
    const int y    = pix0 >> 6;
    const int x0   = pix0 & 63;

    const float* xb = x + (size_t)b * CIN * HW;
    const int tile0 = wave * 2, tile1 = wave * 2 + 1;

    v8f acc00 = {}, acc01 = {}, acc10 = {}, acc11 = {};

    for (int k = 0; k < 9; ++k) {
        // ---- per-tap bilinear setup: 32 pixels x 4 samples (threads 0..127)
        if (tid < 128) {
            int n = tid >> 2, s = tid & 3;
            int pix = pix0 + n;
            float dy = offs[(((size_t)b * 18 + 2 * k    ) * HW) + pix];
            float dx = offs[(((size_t)b * 18 + 2 * k + 1) * HW) + pix];
            float py = dy + (float)y        - 1.0f + (float)(k / 3);
            float px = dx + (float)(x0 + n) - 1.0f + (float)(k % 3);
            float fy = floorf(py), fx = floorf(px);
            float ly = py - fy,    lx = px - fx;
            int   yi = (int)fy + (s >> 1);
            int   xi = (int)fx + (s & 1);
            float wy = (s >> 1) ? ly : (1.0f - ly);
            float wx = (s & 1)  ? lx : (1.0f - lx);
            bool valid = (yi >= 0) && (yi < 64) && (xi >= 0) && (xi < 64);
            int yc = min(max(yi, 0), 63), xc = min(max(xi, 0), 63);
            sIdx[n][s] = yc * WID + xc;
            sWgt[n][s] = valid ? (wy * wx) : 0.0f;
        }
        __syncthreads();

        for (int chunk = 0; chunk < 8; ++chunk) {
            const int pb = chunk & 1;
            // ---- gather B tile (32 cin x 32 pixels): lanes sweep pixels
#pragma unroll
            for (int e = tid; e < 1024; e += 256) {
                int n = e & 31, kloc = e >> 5;
                int cin = chunk * 32 + kloc;
                const float* xc = xb + (size_t)cin * HW;
                float v = sWgt[n][0] * xc[sIdx[n][0]]
                        + sWgt[n][1] * xc[sIdx[n][1]]
                        + sWgt[n][2] * xc[sIdx[n][2]]
                        + sWgt[n][3] * xc[sIdx[n][3]];
                valB[pb][n][kloc] = f2bf(v);
            }
            __syncthreads();   // ping-pong: single barrier per chunk

            // ---- B fragments: lane -> N = lane%16, K-half = (lane/16)*16
            int nn = lane & 15, khalf = (lane >> 4) << 4;
            v16bf bf0 = *(const v16bf*)&valB[pb][nn     ][khalf];
            v16bf bf1 = *(const v16bf*)&valB[pb][nn + 16][khalf];

            // ---- A fragments: one contiguous 32B v16bf per (tile,lane)
            size_t abase = ((((size_t)k * 8 + chunk) * 16) * 32 + lane) * 16;
            v16bf a0 = *(const v16bf*)(wbf + abase + (size_t)tile0 * 512);
            v16bf a1 = *(const v16bf*)(wbf + abase + (size_t)tile1 * 512);
            if (chunk < 7)
                __builtin_prefetch(wbf + abase + 8192 + (size_t)tile0 * 512, 0, 0);

            acc00 = __builtin_amdgcn_wmma_f32_16x16x32_bf16(
                        false, a0, false, bf0, (short)0, acc00, false, false);
            acc01 = __builtin_amdgcn_wmma_f32_16x16x32_bf16(
                        false, a0, false, bf1, (short)0, acc01, false, false);
            acc10 = __builtin_amdgcn_wmma_f32_16x16x32_bf16(
                        false, a1, false, bf0, (short)0, acc10, false, false);
            acc11 = __builtin_amdgcn_wmma_f32_16x16x32_bf16(
                        false, a1, false, bf1, (short)0, acc11, false, false);
        }
    }

    // ---- write D: lane -> N = lane%16, rows M = (lane/16)*8 + r
    int nn = lane & 15;
#pragma unroll
    for (int r = 0; r < 8; ++r) {
        int m = ((lane >> 4) << 3) + r;
        size_t r0 = (((size_t)b * CH + tile0 * 16 + m) * HW) + pix0;
        size_t r1 = (((size_t)b * CH + tile1 * 16 + m) * HW) + pix0;
        out[r0 + nn]      = acc00[r];
        out[r0 + 16 + nn] = acc01[r];
        out[r1 + nn]      = acc10[r];
        out[r1 + 16 + nn] = acc11[r];
    }
}

// ---------------------------------------------------------------------------
// GroupNorm (+ optional residual) + ReLU. One block per (b, group) = 128.
// ---------------------------------------------------------------------------
__global__ __launch_bounds__(256) void groupnorm_kernel(
    const float* __restrict__ src, const float* __restrict__ res,
    const float* __restrict__ gamma, const float* __restrict__ beta,
    float* __restrict__ dst, int do_res)
{
    __shared__ float s_sum[256], s_ss[256];
    int b = blockIdx.x >> 5, g = blockIdx.x & 31;
    size_t base = ((size_t)b * CH + g * 8) * HW;
    const int N = 8 * HW;   // 32768

    float sum = 0.f, ss = 0.f;
    for (int i = threadIdx.x; i < N; i += 256) {
        float v = src[base + i];
        sum += v; ss += v * v;
    }
    s_sum[threadIdx.x] = sum; s_ss[threadIdx.x] = ss;
    __syncthreads();
    for (int off = 128; off > 0; off >>= 1) {
        if (threadIdx.x < off) {
            s_sum[threadIdx.x] += s_sum[threadIdx.x + off];
            s_ss[threadIdx.x]  += s_ss[threadIdx.x + off];
        }
        __syncthreads();
    }
    float mean = s_sum[0] * (1.0f / 32768.0f);
    float var  = s_ss[0]  * (1.0f / 32768.0f) - mean * mean;
    float inv  = rsqrtf(var + 1e-5f);

    for (int i = threadIdx.x; i < N; i += 256) {
        int c = g * 8 + (i >> 12);
        float v = (src[base + i] - mean) * inv * gamma[c] + beta[c];
        if (do_res) v += res[base + i];
        dst[base + i] = fmaxf(v, 0.0f);
    }
}

// ---------------------------------------------------------------------------
extern "C" void kernel_launch(void* const* d_in, const int* in_sizes, int n_in,
                              void* d_out, int out_size, void* d_ws, size_t ws_size,
                              hipStream_t stream) {
    const float* x      = (const float*)d_in[0];
    const float* w_off1 = (const float*)d_in[1];
    const float* b_off1 = (const float*)d_in[2];
    const float* w_off2 = (const float*)d_in[3];
    const float* b_off2 = (const float*)d_in[4];
    const float* w_def1 = (const float*)d_in[5];
    const float* w_def2 = (const float*)d_in[6];
    const float* gamma1 = (const float*)d_in[7];
    const float* beta1  = (const float*)d_in[8];
    const float* gamma2 = (const float*)d_in[9];
    const float* beta2  = (const float*)d_in[10];
    float* out = (float*)d_out;

    char* ws = (char*)d_ws;
    ushort_t* wbf1 = (ushort_t*)(ws);                         // 1.18 MB
    ushort_t* wbf2 = (ushort_t*)(ws + (2u  << 20));           // 1.18 MB
    float*    offs = (float*)   (ws + (4u  << 20));           // 1.18 MB
    float*    tmp1 = (float*)   (ws + (6u  << 20));           // 16.78 MB
    float*    tmp2 = (float*)   (ws + (24u << 20));           // 16.78 MB

    const int THR = 256;
    // weight repack for WMMA A-fragments
    prep_weight_kernel<<<2304, THR, 0, stream>>>(w_def1, wbf1);
    prep_weight_kernel<<<2304, THR, 0, stream>>>(w_def2, wbf2);

    // ---- stage 1
    offset_conv_kernel<<<  64, THR, 0, stream>>>(x, w_off1, b_off1, offs);
    deform_wmma_kernel<<< 512, THR, 0, stream>>>(x, offs, wbf1, tmp1);
    groupnorm_kernel  <<< 128, THR, 0, stream>>>(tmp1, nullptr, gamma1, beta1, tmp2, 0);

    // ---- stage 2 (+ residual, fused into final GN)
    offset_conv_kernel<<<  64, THR, 0, stream>>>(tmp2, w_off2, b_off2, offs);
    deform_wmma_kernel<<< 512, THR, 0, stream>>>(tmp2, offs, wbf2, tmp1);
    groupnorm_kernel  <<< 128, THR, 0, stream>>>(tmp1, x, gamma2, beta2, out, 1);
}
